// EGNN_12000138625538
// MI455X (gfx1250) — matrix operand order repaired and checked by reading, not compiled, not run on hardware
//
#include <hip/hip_runtime.h>
#include <hip/hip_bf16.h>

// ---------------------------------------------------------------------------
// EGNN fused pipeline for gfx1250 (MI455X), wave32 + v_wmma_f32_16x16x32_bf16.
//
// Per step:
//   edge_kernel : gather [h_s|h_r|d2|g] -> phi_e (3 GEMMs) -> atomics to m_i,
//                 phi_x (2 GEMMs + dot) -> atomics to sum_x.  All GEMMs are
//                 16(edge)x64 tiles done with WMMA bf16, one tile per wave.
//   node_kernel : phi_v (2 GEMMs + dot), phi_h (3 GEMMs) + residual,
//                 coord-norm + LayerNorm, writes next node table.
//
// GELU uses the gfx1250 native V_TANH_F32 transcendental so activation math
// runs on the TRANS pipe and co-executes with the WMMA/XDL pipe.
// ---------------------------------------------------------------------------

typedef __attribute__((ext_vector_type(16))) __bf16 v16bf;
typedef __attribute__((ext_vector_type(8)))  float  v8f;

#define NNODES      20000
#define NEDGES      320000
#define HDIM        64
#define GDIM        8
#define NODE_STRIDE 70      // 3 + 3 + H
#define KPAD_E1     160     // 137 padded to 5 K-tiles of 32
#define KPAD_H1     128     // [h | m_i]

// ------------------------------ device helpers -----------------------------

__device__ __forceinline__ float fast_tanh(float x) {
#if __has_builtin(__builtin_amdgcn_tanhf)
    return __builtin_amdgcn_tanhf(x);
#else
    float r;
    asm("v_tanh_f32 %0, %1" : "=v"(r) : "v"(x));
    return r;
#endif
}

__device__ __forceinline__ float gelu_tanh(float x) {
    const float c0 = 0.7978845608028654f;  // sqrt(2/pi)
    float x3 = x * x * x;
    return 0.5f * x * (1.0f + fast_tanh(c0 * (x + 0.044715f * x3)));
}

union FragBF { v16bf v; uint4 q[2]; };

// A fragment (16x32 bf16, M=edge/node row): lanes 0-15 hold M=lane,
// K = k0..k0+7 (q0) and k0+16..k0+23 (q1); lanes 16-31 hold K+8 / K+24.
__device__ __forceinline__ v16bf ldA_frag(const __bf16* A, int kpad, int k0, int lane) {
    int m  = lane & 15;
    int kb = k0 + ((lane >> 4) << 3);
    const __bf16* p = A + m * kpad + kb;
    FragBF f;
    f.q[0] = *(const uint4*)(p);
    f.q[1] = *(const uint4*)(p + 16);
    return f.v;
}

// B fragment (32x16 bf16) from column-major weights W[n][kpad]:
// lanes 0-15: N=lane, K=k0..k0+7 (q0) and k0+16..k0+23 (q1); lanes 16-31: N=lane-16, K+8 / K+24.
__device__ __forceinline__ v16bf ldB_frag(const __bf16* W, int kpad, int k0, int n0, int lane) {
    int n  = n0 + (lane & 15);
    int kb = k0 + ((lane >> 4) << 3);
    const __bf16* p = W + n * kpad + kb;
    FragBF f;
    f.q[0] = *(const uint4*)(p);
    f.q[1] = *(const uint4*)(p + 16);
    return f.v;
}

// 16x64 GEMM tile: A [16 x 32*ktiles] (LDS), W col-major [64 x kpadW] (global).
__device__ __forceinline__ void wmma_gemm(const __bf16* A, int kpadA, int ktiles,
                                          const __bf16* W, int kpadW,
                                          v8f cd[4], int lane) {
#pragma unroll
    for (int j = 0; j < 4; ++j) { v8f z = {0,0,0,0,0,0,0,0}; cd[j] = z; }
    for (int kt = 0; kt < ktiles; ++kt) {
        v16bf a = ldA_frag(A, kpadA, kt * 32, lane);
#pragma unroll
        for (int j = 0; j < 4; ++j) {
            v16bf b = ldB_frag(W, kpadW, kt * 32, j * 16, lane);
            cd[j] = __builtin_amdgcn_wmma_f32_16x16x32_bf16(
                false, a, false, b, (short)0, cd[j], false, false);
        }
    }
}

// C/D layout: VGPR r of lanes 0-15 -> (M=r, N=lane); lanes 16-31 -> (M=8+r, N=lane-16).
// Apply bias (+gelu), store bf16 activation tile back to LDS in A layout.
__device__ __forceinline__ void act_store(__bf16* out, int kpadOut, const v8f cd[4],
                                          const float* __restrict__ bias, bool do_gelu,
                                          int lane) {
    int m0 = (lane >> 4) << 3;
    int nl = lane & 15;
#pragma unroll
    for (int j = 0; j < 4; ++j) {
        int n = j * 16 + nl;
        float bn = bias[n];
#pragma unroll
        for (int r = 0; r < 8; ++r) {
            float v = cd[j][r] + bn;
            if (do_gelu) v = gelu_tanh(v);
            out[(m0 + r) * kpadOut + n] = (__bf16)v;
        }
    }
}

__device__ __forceinline__ float half16_reduce(float p) {
    p += __shfl_xor(p, 1, 32);
    p += __shfl_xor(p, 2, 32);
    p += __shfl_xor(p, 4, 32);
    p += __shfl_xor(p, 8, 32);
    return p;  // reduced within each 16-lane half
}

// ------------------------------ weight prep --------------------------------

__global__ void conv_colmaj(const float* __restrict__ src, __bf16* __restrict__ dst,
                            int K, int Ncols, int kpad) {
    int idx = blockIdx.x * blockDim.x + threadIdx.x;
    int total = Ncols * kpad;
    if (idx >= total) return;
    int n = idx / kpad, k = idx % kpad;
    float v = (k < K) ? src[(size_t)k * Ncols + n] : 0.0f;
    dst[idx] = (__bf16)v;
}

// ------------------------------ edge kernel --------------------------------

struct EdgeP {
    const __bf16 *We1, *We2, *We3, *Wx1, *Wx2;
    const float  *be1, *be2, *be3, *bx1, *bx2, *wx3, *bx3;
};

__global__ __launch_bounds__(256) void egnn_edge_kernel(
    const float* __restrict__ nodes, const float* __restrict__ globals_,
    const int* __restrict__ senders, const int* __restrict__ receivers,
    EdgeP P, float* __restrict__ m_i, float* __restrict__ sum_x, int nTiles) {
    __shared__ __bf16 bufA[8][16 * KPAD_E1];
    __shared__ __bf16 bufB[8][16 * 64];
    __shared__ float  sred[8][16];

    const int tid = threadIdx.x, wave = tid >> 5, lane = tid & 31;
    const int m = lane & 15, part = lane >> 4, mh = part << 3;
    __bf16* bA = bufA[wave];
    __bf16* bB = bufB[wave];

    for (int tile = blockIdx.x * 8 + wave; tile < nTiles; tile += gridDim.x * 8) {
        int e = tile * 16 + m;
        if (e >= NEDGES) e = NEDGES - 1;
        int snd = senders[e], rcv = receivers[e];

        // gather h_s (part 0) / h_r (part 1) into input tile, bf16
        const float* hrow = nodes + (size_t)(part ? rcv : snd) * NODE_STRIDE + 6;
        __bf16* drow = bA + m * KPAD_E1 + part * 64;
#pragma unroll
        for (int q = 0; q < 16; ++q) {
            float4 v4 = *(const float4*)(hrow + q * 4);
            union { __bf16 h[4]; uint2 u; } pk;
            pk.h[0] = (__bf16)v4.x; pk.h[1] = (__bf16)v4.y;
            pk.h[2] = (__bf16)v4.z; pk.h[3] = (__bf16)v4.w;
            *(uint2*)(drow + q * 4) = pk.u;
        }
        const float* xs = nodes + (size_t)snd * NODE_STRIDE;
        const float* xr = nodes + (size_t)rcv * NODE_STRIDE;
        float dx0 = xs[0] - xr[0], dx1 = xs[1] - xr[1], dx2 = xs[2] - xr[2];
        float d2 = dx0 * dx0 + dx1 * dx1 + dx2 * dx2;
        if (part == 0) {
            __bf16* tail = bA + m * KPAD_E1;
            tail[128] = (__bf16)d2;
#pragma unroll
            for (int q = 0; q < GDIM; ++q) tail[129 + q] = (__bf16)globals_[q];
#pragma unroll
            for (int q = 137; q < KPAD_E1; ++q) tail[q] = (__bf16)0.0f;
        }

        int rvv[8];
#pragma unroll
        for (int r = 0; r < 8; ++r) rvv[r] = __shfl(rcv, mh + r, 32);

        v8f cd[4];
        // phi_e: 137->64 (gelu) -> 64 (gelu) -> 64 (linear)
        wmma_gemm(bA, KPAD_E1, 5, P.We1, KPAD_E1, cd, lane);
        act_store(bB, 64, cd, P.be1, true, lane);
        wmma_gemm(bB, 64, 2, P.We2, 64, cd, lane);
        act_store(bA, 64, cd, P.be2, true, lane);
        wmma_gemm(bA, 64, 2, P.We3, 64, cd, lane);

        // m_ij: bf16 copy for phi_x input + atomic scatter into m_i[receiver]
#pragma unroll
        for (int j = 0; j < 4; ++j) {
            int n = j * 16 + m;
            float bn = P.be3[n];
#pragma unroll
            for (int r = 0; r < 8; ++r) {
                float v = cd[j][r] + bn;
                bB[(mh + r) * 64 + n] = (__bf16)v;
                atomicAdd(&m_i[(size_t)rvv[r] * HDIM + n], v);
            }
        }

        // phi_x: 64->64 (gelu) -> 64 (gelu) -> 1 (linear, done as dot)
        wmma_gemm(bB, 64, 2, P.Wx1, 64, cd, lane);
        act_store(bA, 64, cd, P.bx1, true, lane);
        wmma_gemm(bA, 64, 2, P.Wx2, 64, cd, lane);
#pragma unroll
        for (int r = 0; r < 8; ++r) {
            float p = 0.0f;
#pragma unroll
            for (int j = 0; j < 4; ++j) {
                int n = j * 16 + m;
                p += gelu_tanh(cd[j][r] + P.bx2[n]) * P.wx3[n];
            }
            p = half16_reduce(p);
            if (m == 0) sred[wave][mh + r] = p;
        }
        __builtin_amdgcn_wave_barrier();
        if (part == 0) {
            float s = sred[wave][m] + P.bx3[0];
            atomicAdd(&sum_x[(size_t)rcv * 3 + 0], dx0 * s);
            atomicAdd(&sum_x[(size_t)rcv * 3 + 1], dx1 * s);
            atomicAdd(&sum_x[(size_t)rcv * 3 + 2], dx2 * s);
        }
        __builtin_amdgcn_wave_barrier();
    }
}

// ------------------------------ node kernel --------------------------------

struct NodeP {
    const __bf16 *Wv1, *Wv2, *Wh1, *Wh2, *Wh3;
    const float  *bv1, *bv2, *wv3, *bv3, *bh1, *bh2, *bh3;
    const float  *lns, *lnb, *xscale, *vscale;
};

__global__ __launch_bounds__(128) void egnn_node_kernel(
    const float* __restrict__ nodes_in, const float* __restrict__ m_i,
    const float* __restrict__ sum_x, NodeP P, float* __restrict__ nodes_out,
    int nTiles) {
    __shared__ __bf16 bufA[4][16 * KPAD_H1];
    __shared__ __bf16 bufB[4][16 * 64];
    __shared__ float  sredv[4][16];
    __shared__ float  smu[4][16];
    __shared__ float  svar[4][16];

    const int tid = threadIdx.x, wave = tid >> 5, lane = tid & 31;
    const int m = lane & 15, part = lane >> 4, mh = part << 3;
    __bf16* bA = bufA[wave];
    __bf16* bB = bufB[wave];

    for (int tile = blockIdx.x * 4 + wave; tile < nTiles; tile += gridDim.x * 4) {
        int i = tile * 16 + m;
        if (i >= NNODES) i = NNODES - 1;

        // stage [h | m_i] row (part 0: h, part 1: m_i)
        const float* src = part ? (m_i + (size_t)i * HDIM)
                                : (nodes_in + (size_t)i * NODE_STRIDE + 6);
        __bf16* drow = bA + m * KPAD_H1 + part * 64;
#pragma unroll
        for (int q = 0; q < 16; ++q) {
            float4 v4 = *(const float4*)(src + q * 4);
            union { __bf16 h[4]; uint2 u; } pk;
            pk.h[0] = (__bf16)v4.x; pk.h[1] = (__bf16)v4.y;
            pk.h[2] = (__bf16)v4.z; pk.h[3] = (__bf16)v4.w;
            *(uint2*)(drow + q * 4) = pk.u;
        }

        v8f cd[4];
        // phi_v: 64->64 (gelu) -> 64 (gelu) -> 1 (dot)
        wmma_gemm(bA, KPAD_H1, 2, P.Wv1, 64, cd, lane);
        act_store(bB, 64, cd, P.bv1, true, lane);
        wmma_gemm(bB, 64, 2, P.Wv2, 64, cd, lane);
#pragma unroll
        for (int r = 0; r < 8; ++r) {
            float p = 0.0f;
#pragma unroll
            for (int j = 0; j < 4; ++j) {
                int n = j * 16 + m;
                p += gelu_tanh(cd[j][r] + P.bv2[n]) * P.wv3[n];
            }
            p = half16_reduce(p);
            if (m == 0) sredv[wave][mh + r] = p;
        }
        __builtin_amdgcn_wave_barrier();
        float sv = sredv[wave][m] + P.bv3[0];

        // phi_h: 128->64 (gelu) -> 64 (gelu) -> 64 (linear) + residual h
        wmma_gemm(bA, KPAD_H1, 4, P.Wh1, KPAD_H1, cd, lane);
        act_store(bB, 64, cd, P.bh1, true, lane);
        wmma_gemm(bB, 64, 2, P.Wh2, 64, cd, lane);
        act_store(bA + 64, KPAD_H1, cd, P.bh2, true, lane);  // reuse m_i half
        wmma_gemm(bA + 64, KPAD_H1, 2, P.Wh3, 64, cd, lane);

        float hv[4][8];
#pragma unroll
        for (int j = 0; j < 4; ++j) {
            int n = j * 16 + m;
            float bn = P.bh3[n];
#pragma unroll
            for (int r = 0; r < 8; ++r) {
                int row = tile * 16 + mh + r;
                if (row >= NNODES) row = NNODES - 1;
                float h0 = nodes_in[(size_t)row * NODE_STRIDE + 6 + n];
                hv[j][r] = cd[j][r] + bn + h0;
            }
        }
        // LayerNorm stats across N (feature) dimension
#pragma unroll
        for (int r = 0; r < 8; ++r) {
            float p1 = 0.0f, p2 = 0.0f;
#pragma unroll
            for (int j = 0; j < 4; ++j) { float v = hv[j][r]; p1 += v; p2 += v * v; }
            p1 = half16_reduce(p1);
            p2 = half16_reduce(p2);
            if (m == 0) { smu[wave][mh + r] = p1; svar[wave][mh + r] = p2; }
        }
        __builtin_amdgcn_wave_barrier();
#pragma unroll
        for (int r = 0; r < 8; ++r) {
            float mu  = smu[wave][mh + r] * (1.0f / HDIM);
            float var = svar[wave][mh + r] * (1.0f / HDIM) - mu * mu;
            float rs  = rsqrtf(var + 1e-6f);
            int row = tile * 16 + mh + r;
            if (row >= NNODES) row = NNODES - 1;
#pragma unroll
            for (int j = 0; j < 4; ++j) {
                int n = j * 16 + m;
                nodes_out[(size_t)row * NODE_STRIDE + 6 + n] =
                    (hv[j][r] - mu) * rs * P.lns[n] + P.lnb[n];
            }
        }

        // x / v update + coord norms (one lane per node)
        if (part == 0) {
            const float* nr = nodes_in + (size_t)i * NODE_STRIDE;
            float x0 = nr[0], x1 = nr[1], x2 = nr[2];
            float v0 = nr[3], v1 = nr[4], v2 = nr[5];
            const float invEm1 = 1.0f / (float)(NEDGES - 1);
            float vp0 = sum_x[(size_t)i * 3 + 0] * invEm1 + sv * v0;
            float vp1 = sum_x[(size_t)i * 3 + 1] * invEm1 + sv * v1;
            float vp2 = sum_x[(size_t)i * 3 + 2] * invEm1 + sv * v2;
            float xp0 = x0 + vp0, xp1 = x1 + vp1, xp2 = x2 + vp2;
            float xn = sqrtf(xp0 * xp0 + xp1 * xp1 + xp2 * xp2);
            float xsc = P.xscale[0] / fmaxf(xn, 1e-8f);
            xp0 *= xsc; xp1 *= xsc; xp2 *= xsc;
            float vn = sqrtf(vp0 * vp0 + vp1 * vp1 + vp2 * vp2);
            float vsc = P.vscale[0] / fmaxf(vn, 1e-8f);
            vp0 *= vsc; vp1 *= vsc; vp2 *= vsc;
            float* outr = nodes_out + (size_t)i * NODE_STRIDE;
            outr[0] = xp0; outr[1] = xp1; outr[2] = xp2;
            outr[3] = vp0; outr[4] = vp1; outr[5] = vp2;
        }
        __builtin_amdgcn_wave_barrier();
    }
}

// ------------------------------ host launcher ------------------------------
//
// Input order (JAX sorted-dict flattening): d_in[0]=nodes, [1]=globals_,
// [2]=senders, [3]=receivers, then per step (28 leaves):
//  +0 ln_bias +1 ln_scale
//  +2..4 phi_e.W0/1/2   +5..7  phi_e.b0/1/2
//  +8..10 phi_h.W0/1/2  +11..13 phi_h.b0/1/2
//  +14..16 phi_v.W0/1/2 +17..19 phi_v.b0/1/2
//  +20..22 phi_x.W0/1/2 +23..25 phi_x.b0/1/2
//  +26 v_scale +27 x_scale

extern "C" void kernel_launch(void* const* d_in, const int* in_sizes, int n_in,
                              void* d_out, int out_size, void* d_ws, size_t ws_size,
                              hipStream_t stream) {
    (void)in_sizes; (void)n_in; (void)out_size; (void)ws_size;

    const float* nodes0   = (const float*)d_in[0];
    const float* globals_ = (const float*)d_in[1];
    const int*   senders  = (const int*)d_in[2];
    const int*   receivers= (const int*)d_in[3];

    // workspace layout
    const size_t STEP_HALVES = 51200;  // bf16 weight blob per step
    char* ws = (char*)d_ws;
    __bf16* blob  = (__bf16*)ws;
    size_t  off   = 4 * STEP_HALVES * sizeof(__bf16);        // 409600
    float*  sum_x = (float*)(ws + off);   off += (size_t)NNODES * 3 * 4;       // 240000
    float*  m_i   = (float*)(ws + off);   off += (size_t)NNODES * HDIM * 4;    // 5.12MB
    float*  nodesA= (float*)(ws + off);   off += (size_t)NNODES * NODE_STRIDE * 4;
    float*  nodesB= (float*)(ws + off);

    // per-step weight offsets inside the blob (halves)
    const size_t oWe1 = 0, oWe2 = 10240, oWe3 = 14336, oWx1 = 18432, oWx2 = 22528,
                 oWv1 = 26624, oWv2 = 30720, oWh1 = 34816, oWh2 = 43008, oWh3 = 47104;

    // convert all weights up front (column-major bf16, zero-padded K)
    for (int s = 0; s < 4; ++s) {
        int base = 4 + s * 28;
        __bf16* b = blob + s * STEP_HALVES;
        struct { int idx; size_t dst; int K, kpad; } cv[10] = {
            { base + 2,  oWe1, 137, KPAD_E1 }, { base + 3,  oWe2, 64, 64 },
            { base + 4,  oWe3, 64, 64 },       { base + 20, oWx1, 64, 64 },
            { base + 21, oWx2, 64, 64 },       { base + 14, oWv1, 64, 64 },
            { base + 15, oWv2, 64, 64 },       { base + 8,  oWh1, 128, KPAD_H1 },
            { base + 9,  oWh2, 64, 64 },       { base + 10, oWh3, 64, 64 },
        };
        for (int c = 0; c < 10; ++c) {
            int total = 64 * cv[c].kpad;
            conv_colmaj<<<(total + 255) / 256, 256, 0, stream>>>(
                (const float*)d_in[cv[c].idx], b + cv[c].dst, cv[c].K, 64, cv[c].kpad);
        }
    }

    const int nTilesE = NEDGES / 16;               // 20000
    const int nTilesN = (NNODES + 15) / 16;        // 1250
    const int edgeBlocks = (nTilesE + 7) / 8;
    const int nodeBlocks = (nTilesN + 3) / 4;

    for (int s = 0; s < 4; ++s) {
        int base = 4 + s * 28;
        __bf16* b = blob + s * STEP_HALVES;

        EdgeP ep;
        ep.We1 = b + oWe1; ep.We2 = b + oWe2; ep.We3 = b + oWe3;
        ep.Wx1 = b + oWx1; ep.Wx2 = b + oWx2;
        ep.be1 = (const float*)d_in[base + 5];
        ep.be2 = (const float*)d_in[base + 6];
        ep.be3 = (const float*)d_in[base + 7];
        ep.bx1 = (const float*)d_in[base + 23];
        ep.bx2 = (const float*)d_in[base + 24];
        ep.bx3 = (const float*)d_in[base + 25];
        ep.wx3 = (const float*)d_in[base + 22];

        NodeP np;
        np.Wv1 = b + oWv1; np.Wv2 = b + oWv2;
        np.Wh1 = b + oWh1; np.Wh2 = b + oWh2; np.Wh3 = b + oWh3;
        np.bv1 = (const float*)d_in[base + 17];
        np.bv2 = (const float*)d_in[base + 18];
        np.bv3 = (const float*)d_in[base + 19];
        np.wv3 = (const float*)d_in[base + 16];
        np.bh1 = (const float*)d_in[base + 11];
        np.bh2 = (const float*)d_in[base + 12];
        np.bh3 = (const float*)d_in[base + 13];
        np.lnb = (const float*)d_in[base + 0];
        np.lns = (const float*)d_in[base + 1];
        np.vscale = (const float*)d_in[base + 26];
        np.xscale = (const float*)d_in[base + 27];

        const float* nin  = (s == 0) ? nodes0 : ((s == 2) ? nodesB : nodesA);
        float*       nout = (s == 3) ? (float*)d_out : ((s & 1) ? nodesB : nodesA);

        hipMemsetAsync(sum_x, 0, (size_t)NNODES * 3 * 4, stream);
        hipMemsetAsync(m_i,   0, (size_t)NNODES * HDIM * 4, stream);

        egnn_edge_kernel<<<edgeBlocks, 256, 0, stream>>>(
            nin, globals_, senders, receivers, ep, m_i, sum_x, nTilesE);
        egnn_node_kernel<<<nodeBlocks, 128, 0, stream>>>(
            nin, m_i, sum_x, np, nout, nTilesN);
    }
}
